// HybridRecPromptModel_14328010899691
// MI455X (gfx1250) — matrix operand order repaired and checked by reading, not compiled, not run on hardware
//
#include <hip/hip_runtime.h>

typedef __attribute__((ext_vector_type(8)))  __bf16 v8bf;
typedef __attribute__((ext_vector_type(16))) __bf16 v16bf;
typedef __attribute__((ext_vector_type(8)))  float  v8f;
typedef __attribute__((ext_vector_type(4)))  float  f4v;

union FragBF { v16bf v; v8bf h[2]; };

#define NUM_ITEMS 100000
#define D_SEM     768
#define D_PROMPT  384
#define K_TOT     1152
#define D_MODEL   2048
#define M_TOT     12800   // B*T = 64*200

#define BM   128
#define BN   128
#define KC   32
#define LDSS 40           // 32 bf16 + 8 pad -> 80B row stride

__device__ __forceinline__ v8bf cvt8(f4v a, f4v b) {
    v8bf r;
    r[0]=(__bf16)a[0]; r[1]=(__bf16)a[1]; r[2]=(__bf16)a[2]; r[3]=(__bf16)a[3];
    r[4]=(__bf16)b[0]; r[5]=(__bf16)b[1]; r[6]=(__bf16)b[2]; r[7]=(__bf16)b[3];
    return r;
}

struct Stage { v8bf a0, a1, b0, b1; };

__global__ __launch_bounds__(256)
void gemm_gather_kernel(const int* __restrict__ ids,
                        const float* __restrict__ E_sem,
                        const float* __restrict__ v_prompt,
                        const float* __restrict__ W,     // [D_MODEL, K_TOT] row-major
                        float* __restrict__ out)         // [M_TOT, D_MODEL]
{
    __shared__ __align__(16) __bf16 sA[2][BM * LDSS];
    __shared__ __align__(16) __bf16 sB[2][BN * LDSS];

    const int m0   = blockIdx.x * BM;
    const int n0   = blockIdx.y * BN;
    const int t    = threadIdx.x;
    const int lane = t & 31;
    const int wave = t >> 5;
    const int wm   = wave & 3;    // 0..3 -> 32-row strip
    const int wn   = wave >> 2;   // 0..1 -> 64-col strip
    const int h    = lane >> 4;   // fragment K-half selector
    const int l16  = lane & 15;

    // ---- cooperative loaders: 4 threads/row, 8 contiguous K each; 2 rows/thread ----
    const int r0 = t >> 2;            // 0..63
    const int r1 = r0 + 64;           // 64..127
    const int lk = (t & 3) * 8;       // 0,8,16,24

    const int  id0 = ids[m0 + r0];
    const int  id1 = ids[m0 + r1];
    const bool valid0 = (id0 >= 0) && (id0 < NUM_ITEMS);
    const bool valid1 = (id1 >= 0) && (id1 < NUM_ITEMS);
    const long sid0 = valid0 ? (long)id0 : 0;   // clamp for safe branchless gather
    const long sid1 = valid1 ? (long)id1 : 0;
    const float* es0 = E_sem    + sid0 * D_SEM;
    const float* es1 = E_sem    + sid1 * D_SEM;
    const float* vp0 = v_prompt + sid0 * D_PROMPT - D_SEM;   // indexed by kg directly
    const float* vp1 = v_prompt + sid1 * D_PROMPT - D_SEM;
    const float* w0  = W + (long)(n0 + r0) * K_TOT;
    const float* w1  = W + (long)(n0 + r1) * K_TOT;

    auto loadGlobal = [&](int k0) -> Stage {
        const int kg = k0 + lk;
        const bool sem = kg < D_SEM;                // 768 % KC == 0: no straddle
        const float* pa0 = (sem ? es0 : vp0) + kg;
        const float* pa1 = (sem ? es1 : vp1) + kg;
        f4v x0 = *(const f4v*)(pa0);
        f4v x1 = *(const f4v*)(pa0 + 4);
        f4v y0 = *(const f4v*)(pa1);
        f4v y1 = *(const f4v*)(pa1 + 4);
        const f4v z = {0.f, 0.f, 0.f, 0.f};
        if (!valid0) { x0 = z; x1 = z; }            // masked rows -> exact 0 output
        if (!valid1) { y0 = z; y1 = z; }
        Stage st;
        st.a0 = cvt8(x0, x1);
        st.a1 = cvt8(y0, y1);
        st.b0 = cvt8(*(const f4v*)(w0 + kg), *(const f4v*)(w0 + kg + 4));
        st.b1 = cvt8(*(const f4v*)(w1 + kg), *(const f4v*)(w1 + kg + 4));
        return st;
    };

    auto storeLDS = [&](int buf, const Stage& st) {
        *(v8bf*)&sA[buf][r0 * LDSS + lk] = st.a0;
        *(v8bf*)&sA[buf][r1 * LDSS + lk] = st.a1;
        *(v8bf*)&sB[buf][r0 * LDSS + lk] = st.b0;
        *(v8bf*)&sB[buf][r1 * LDSS + lk] = st.b1;
    };

    v8f acc[2][4];
    #pragma unroll
    for (int mi = 0; mi < 2; ++mi)
        #pragma unroll
        for (int ni = 0; ni < 4; ++ni)
            acc[mi][ni] = (v8f){};

    // ---- software pipeline: global->reg prefetch overlaps WMMA; 1 barrier/iter ----
    Stage st = loadGlobal(0);
    storeLDS(0, st);
    __syncthreads();

    int buf = 0;
    #pragma unroll 1
    for (int k0 = 0; k0 < K_TOT; k0 += KC) {
        const bool more = (k0 + KC) < K_TOT;
        Stage nx;
        if (more) nx = loadGlobal(k0 + KC);

        // A fragments: row m = wm*32 + mi*16 + lane%16; K per ISA 16-bit layout
        FragBF fa[2];
        #pragma unroll
        for (int mi = 0; mi < 2; ++mi) {
            const int ml = (wm * 32 + mi * 16 + l16) * LDSS;
            fa[mi].h[0] = *(const v8bf*)&sA[buf][ml + 8 * h];        // K = 8h..8h+7
            fa[mi].h[1] = *(const v8bf*)&sA[buf][ml + 16 + 8 * h];   // K = 16+8h..+7
        }
        // B fragments: col n = wn*64 + ni*16 + lane%16; K per ISA 16-bit layout
        FragBF fb[4];
        #pragma unroll
        for (int ni = 0; ni < 4; ++ni) {
            const int nl = (wn * 64 + ni * 16 + l16) * LDSS;
            fb[ni].h[0] = *(const v8bf*)&sB[buf][nl + 8 * h];
            fb[ni].h[1] = *(const v8bf*)&sB[buf][nl + 16 + 8 * h];
        }
        #pragma unroll
        for (int mi = 0; mi < 2; ++mi)
            #pragma unroll
            for (int ni = 0; ni < 4; ++ni)
                acc[mi][ni] = __builtin_amdgcn_wmma_f32_16x16x32_bf16(
                    false, fa[mi].v, false, fb[ni].v,
                    (short)0, acc[mi][ni], false, false);

        if (more) storeLDS(buf ^ 1, nx);
        __syncthreads();
        buf ^= 1;
    }

    // ---- store C: VGPR r of each 16x16 f32 frag holds (m = r + 8h, n = lane%16) ----
    const long mbase = m0 + wm * 32 + 8 * h;
    const int  nbase = n0 + wn * 64 + l16;
    #pragma unroll
    for (int mi = 0; mi < 2; ++mi) {
        #pragma unroll
        for (int r = 0; r < 8; ++r) {
            float* orow = out + (mbase + mi * 16 + r) * (long)D_MODEL + nbase;
            #pragma unroll
            for (int ni = 0; ni < 4; ++ni)
                orow[ni * 16] = acc[mi][ni][r];
        }
    }
}

__global__ __launch_bounds__(256)
void mask_tok_kernel(const int* __restrict__ ids,
                     int* __restrict__ amask,
                     int* __restrict__ tok)
{
    const int i = blockIdx.x * 256 + threadIdx.x;
    if (i < M_TOT) {
        const int id = ids[i];
        const bool v = (id >= 0) && (id < NUM_ITEMS);
        amask[i] = v ? 1 : 0;
        tok[i]   = v ? id : -1;
    }
}

extern "C" void kernel_launch(void* const* d_in, const int* in_sizes, int n_in,
                              void* d_out, int out_size, void* d_ws, size_t ws_size,
                              hipStream_t stream) {
    (void)in_sizes; (void)n_in; (void)out_size; (void)d_ws; (void)ws_size;
    const int*   ids      = (const int*)d_in[0];
    // d_in[1] = lengths (already folded into ids padding; unused)
    const float* E_sem    = (const float*)d_in[2];
    const float* v_prompt = (const float*)d_in[3];
    const float* proj_w   = (const float*)d_in[4];

    float* embeds = (float*)d_out;
    int*   amask  = (int*)d_out + (size_t)M_TOT * D_MODEL;
    int*   tok    = amask + M_TOT;

    dim3 grid(M_TOT / BM, D_MODEL / BN);
    gemm_gather_kernel<<<grid, 256, 0, stream>>>(ids, E_sem, v_prompt, proj_w, embeds);
    mask_tok_kernel<<<(M_TOT + 255) / 256, 256, 0, stream>>>(ids, amask, tok);
}
